// RecursiveNeuronModel_46256797778326
// MI455X (gfx1250) — matrix operand order repaired and checked by reading, not compiled, not run on hardware
//
#include <hip/hip_runtime.h>
#include <hip/hip_bf16.h>
#include <cstdint>

// ---------------- problem constants ----------------
#define B_    8
#define T_    4096
#define NB_   4
#define SL_   128
#define SB_   128
#define COUT  16
#define KW    9
#define NSYN  1536
#define TROWS (T_ + 24)          // 12 zero guard rows each side

// ---------------- tiling constants -----------------
#define TB     128               // output time columns per block
#define XP     392               // x LDS pitch in bf16 (384 ch + 8 pad)
#define XLOAD  152               // x rows actually loaded (TB + 24)
#define XROWS  160               // x rows allocated (slack for overshoot reads)
#define LP     40                // leaves LDS pitch (32 ch + 8 pad)
#define LROWS  152               // leaves rows allocated (144 used + slack)
#define HP     168               // hb|inter LDS pitch (160 ch + 8 pad)
#define HROWS  144               // hb rows allocated (136 used + slack)

typedef __attribute__((ext_vector_type(16))) __bf16 v16bf;
typedef __attribute__((ext_vector_type(8)))  __bf16 v8bf;
typedef __attribute__((ext_vector_type(8)))  float  v8f;
typedef __attribute__((ext_vector_type(4)))  unsigned int u32x4;
typedef __attribute__((ext_vector_type(8)))  int i32x8;
typedef __attribute__((ext_vector_type(4)))  int i32x4;

#define SX_ELTS (XROWS * XP)
#define SLV_ELTS (LROWS * LP)
#define SH_ELTS (HROWS * HP)
#define SMEM_BYTES ((SX_ELTS + SLV_ELTS + SH_ELTS) * 2)

// =====================================================================
// Gather + permute + f32->bf16 convert into branch-contiguous layout:
// xg[b][nb][row][c], row = t + 12 (12 zero guard rows both ends),
// c: 0..127 leaf0, 128..255 leaf1, 256..383 branch-leaf, 384..391 zero pad.
// =====================================================================
__global__ void k_gather(const float* __restrict__ x,
                         const int* __restrict__ idx_leaf,
                         const int* __restrict__ idx_branch,
                         __bf16* __restrict__ xg, long total)
{
    long i = (long)blockIdx.x * blockDim.x + threadIdx.x;
    if (i >= total) return;
    int  c   = (int)(i % XP);
    long r   = i / XP;
    int  row = (int)(r % TROWS);
    long q   = r / TROWS;
    int  nb  = (int)(q % NB_);
    int  b   = (int)(q / NB_);
    int  t   = row - 12;
    float v = 0.f;
    if (c < 384 && t >= 0 && t < T_) {
        int src;
        if (c < 256) src = idx_leaf[(nb * 2 + (c >> 7)) * SL_ + (c & 127)];
        else         src = idx_branch[nb * SB_ + (c - 256)];
        v = x[((long)b * NSYN + src) * T_ + t];
    }
    xg[i] = (__bf16)v;
}

// =====================================================================
// Pack conv weights [nT][Cout][Cin][KW] (f32) into WMMA A-fragment
// layout (bf16): out[((tn*Mt+mt)*KW + k)*NC + cc][lane][e], so a wave
// loads its whole 16x32 A tile with one 32B load per lane.
// e -> ci = cc*32 + hk*16 + e (contiguous k order, same as the B reads)
// Cin may be < NC*32 (zero-padded, e.g. w_br 144 -> 160).
// =====================================================================
__global__ void k_pack(const float* __restrict__ w, __bf16* __restrict__ out,
                       int nT, int Cout, int Cin, int NC)
{
    int i = blockIdx.x * blockDim.x + threadIdx.x;
    int Mt = Cout >> 4;
    int total = nT * Mt * KW * NC * 32;
    if (i >= total) return;
    int lane = i & 31; int f = i >> 5;
    int cc = f % NC;  f /= NC;
    int k  = f % KW;  f /= KW;
    int mt = f % Mt;  int tn = f / Mt;
    int m = lane & 15, hk = lane >> 4;
    long obase = (long)i * 16;
    for (int e = 0; e < 16; ++e) {
        int kk = hk * 16 + e;
        int ci = cc * 32 + kk;
        float v = 0.f;
        if (ci < Cin)
            v = w[(((long)tn * Cout + (mt * 16 + m)) * Cin + ci) * KW + k];
        out[obase + e] = (__bf16)v;
    }
}

// =====================================================================
// Fused branch kernel: one block = (batch b, branch nb, 128-col t-block).
// LDS (time-major, bf16):
//   sx[160][392]  x tile rows t0-12 .. t0+139 (TDM-loaded)
//   sl[152][40]   leaves (ch 0..31), rows t0-8 .. t0+135
//   sh[144][168]  hb (ch 0..127) | inter (128..143) | zeros (144..159)
// Phases: TDM load -> leaves+hb WMMA -> inter WMMA -> br WMMA -> feats.
// =====================================================================
__global__ __launch_bounds__(256, 1)
void k_branch(const __bf16* __restrict__ xg,
              const __bf16* __restrict__ pwL, const float* __restrict__ bL,
              const __bf16* __restrict__ pwI, const float* __restrict__ bI,
              const __bf16* __restrict__ pwH, const float* __restrict__ bH,
              const __bf16* __restrict__ pwB, const float* __restrict__ bB,
              float* __restrict__ feats)
{
    extern __shared__ __bf16 smem[];
    __bf16* sx = smem;
    __bf16* sl = smem + SX_ELTS;
    __bf16* sh = smem + SX_ELTS + SLV_ELTS;

    const int blk = blockIdx.x;
    const int tb = blk & 31;
    const int nb = (blk >> 5) & 3;
    const int b  = blk >> 7;
    const int t0 = tb * TB;
    const int tid  = threadIdx.x;
    const int wave = tid >> 5;
    const int lane = tid & 31;
    const int n  = lane & 15;      // N index of B/C fragments
    const int hk = lane >> 4;      // K/M half select

    // zero channel columns 144..159 of sh (padding channels of br conv)
    for (int i = tid; i < HROWS * 16; i += 256) {
        int r = i >> 4, c = 144 + (i & 15);
        sh[r * HP + c] = (__bf16)0.f;
    }

    // ---- Phase A: TDM 2-D tile load, 152 rows x 392 bf16, into sx ----
    if (wave == 0) {
        const __bf16* src = xg + ((long)(b * NB_ + nb) * TROWS + t0) * XP;
        unsigned long long ga = (unsigned long long)(uintptr_t)src;
        u32x4 g0;
        g0.x = 1u;                                      // count = 1 (valid)
        g0.y = 0u;                                      // lds_addr (sx at offset 0)
        g0.z = (unsigned)(ga & 0xffffffffull);          // global_addr[31:0]
        g0.w = (unsigned)((ga >> 32) & 0x1ffffffull)    // global_addr[56:32]
             | (2u << 30);                              // type = 2 (image)
        i32x8 g1;
        g1[0] = 0x00010000;                             // data_size = 1 (2 bytes)
        g1[1] = (XP << 16);                             // tensor_dim0[15:0]
        g1[2] = (XLOAD << 16);                          // tensor_dim1[15:0]
        g1[3] = (XP << 16);                             // tile_dim0
        g1[4] = XLOAD;                                  // tile_dim1 (tile_dim2=0 -> 2D)
        g1[5] = XP;                                     // tensor_dim0_stride[31:0]
        g1[6] = 0;
        g1[7] = 0;
        i32x4 z4 = {0, 0, 0, 0};
#if defined(__clang_major__) && (__clang_major__ >= 23)
        i32x8 z8 = {0, 0, 0, 0, 0, 0, 0, 0};
        __builtin_amdgcn_tensor_load_to_lds(g0, g1, z4, z4, z8, 0);
#else
        __builtin_amdgcn_tensor_load_to_lds(g0, g1, z4, z4, 0);
#endif
        __builtin_amdgcn_s_wait_tensorcnt(0);
    }
    __syncthreads();

    // ---- Phase B: hb (8 Mtiles x 9 Ntiles) + leaves (2 x 9 Ntiles) ----
    for (int task = wave; task < 90; task += 8) {
        const __bf16* pw;
        const float*  bias;
        int xrAdd, cib, colOff, rowLimit, nt;
        __bf16* dst; int pitch;
        if (task < 72) {                           // hb: 128 <- x[256..383]
            int mt = task / 9; nt = task % 9;
            pw   = pwH + ((long)(nb * 8 + mt) * KW) * 4 * 512;
            bias = bH + nb * 128 + mt * 16;
            xrAdd = 4; cib = 256;
            dst = sh; pitch = HP; colOff = mt * 16; rowLimit = 136;
        } else {                                   // leaf li: 16 <- x[li*128..]
            int t2 = task - 72; int li = t2 / 9; nt = t2 % 9;
            pw   = pwL + ((long)(nb * 2 + li) * KW) * 4 * 512;
            bias = bL + (nb * 2 + li) * 16;
            xrAdd = 0; cib = li * 128;
            dst = sl; pitch = LP; colOff = li * 16; rowLimit = 144;
        }
        v8f acc = {0.f, 0.f, 0.f, 0.f, 0.f, 0.f, 0.f, 0.f};
        const int xrBase = nt * 16 + n + xrAdd;
        #pragma unroll
        for (int k = 0; k < KW; ++k) {
            const __bf16* brow = sx + (long)(xrBase + k) * XP + cib + hk * 16;
            const __bf16* arow = pw + (long)k * 4 * 512 + lane * 16;
            #pragma unroll
            for (int cc = 0; cc < 4; ++cc) {
                v16bf A  = *(const v16bf*)(arow + cc * 512);
                v16bf Bm = *(const v16bf*)(brow + cc * 32);
                acc = __builtin_amdgcn_wmma_f32_16x16x32_bf16(
                          false, A, false, Bm, (short)0, acc, false, false);
            }
        }
        int outRow = nt * 16 + n;
        if (outRow < rowLimit) {
            v8bf o;
            #pragma unroll
            for (int r = 0; r < 8; ++r) {
                float v = acc[r] + bias[hk * 8 + r];
                v = v > 0.f ? v : v * 0.25f;
                o[r] = (__bf16)v;
            }
            *(v8bf*)(dst + (long)outRow * pitch + colOff + hk * 8) = o;
        }
    }
    __syncthreads();

    // ---- Phase C: inter conv 16 <- leaves(32), into sh[ch 128..143] ----
    for (int nt = wave; nt < 9; nt += 8) {
        v8f acc = {0.f, 0.f, 0.f, 0.f, 0.f, 0.f, 0.f, 0.f};
        #pragma unroll
        for (int k = 0; k < KW; ++k) {
            v16bf A  = *(const v16bf*)(pwI + ((long)nb * KW + k) * 512 + lane * 16);
            v16bf Bm = *(const v16bf*)(sl + (long)(nt * 16 + n + k) * LP + hk * 16);
            acc = __builtin_amdgcn_wmma_f32_16x16x32_bf16(
                      false, A, false, Bm, (short)0, acc, false, false);
        }
        int outRow = nt * 16 + n;
        if (outRow < 136) {
            v8bf o;
            #pragma unroll
            for (int r = 0; r < 8; ++r) {
                float v = acc[r] + bI[nb * 16 + hk * 8 + r];
                v = v > 0.f ? v : v * 0.25f;
                o[r] = (__bf16)v;
            }
            *(v8bf*)(sh + (long)outRow * HP + 128 + hk * 8) = o;
        }
    }
    __syncthreads();

    // ---- Phase D: br conv 16 <- [hb|inter](144->160), write feats -----
    for (int nt = wave; nt < 8; nt += 8) {
        v8f acc = {0.f, 0.f, 0.f, 0.f, 0.f, 0.f, 0.f, 0.f};
        #pragma unroll
        for (int k = 0; k < KW; ++k) {
            const __bf16* brow = sh + (long)(nt * 16 + n + k) * HP + hk * 16;
            const __bf16* arow = pwB + (((long)nb * KW + k) * 5) * 512 + lane * 16;
            #pragma unroll
            for (int cc = 0; cc < 5; ++cc) {
                v16bf A  = *(const v16bf*)(arow + cc * 512);
                v16bf Bm = *(const v16bf*)(brow + cc * 32);
                acc = __builtin_amdgcn_wmma_f32_16x16x32_bf16(
                          false, A, false, Bm, (short)0, acc, false, false);
            }
        }
        int tcol = t0 + nt * 16 + n;
        #pragma unroll
        for (int r = 0; r < 8; ++r) {
            float v = acc[r] + bB[nb * 16 + hk * 8 + r];
            v = v > 0.f ? v : v * 0.25f;
            feats[((long)b * 64 + nb * 16 + hk * 8 + r) * T_ + tcol] = v;
        }
    }
}

// =====================================================================
// Head: s/v 1-channel convs over feats (B,64,T). 77 MFLOP -> scalar VALU.
// =====================================================================
__global__ void k_head(const float* __restrict__ feats,
                       const float* __restrict__ ws, const float* __restrict__ bs,
                       const float* __restrict__ wv, const float* __restrict__ bv,
                       float* __restrict__ out)
{
    int i = blockIdx.x * blockDim.x + threadIdx.x;
    if (i >= B_ * T_) return;
    int t = i & (T_ - 1);
    int b = i >> 12;
    float s = bs[0], v = bv[0];
    for (int c = 0; c < 64; ++c) {
        const float* f = feats + ((long)b * 64 + c) * T_;
        #pragma unroll
        for (int k = 0; k < KW; ++k) {
            int tt = t + k - 4;
            if (tt >= 0 && tt < T_) {
                float xv = f[tt];
                s += ws[c * KW + k] * xv;
                v += wv[c * KW + k] * xv;
            }
        }
    }
    out[i] = s;
    out[B_ * T_ + i] = v;
}

// =====================================================================
extern "C" void kernel_launch(void* const* d_in, const int* in_sizes, int n_in,
                              void* d_out, int out_size, void* d_ws, size_t ws_size,
                              hipStream_t stream)
{
    const float* x        = (const float*)d_in[0];
    const int*   idx_leaf = (const int*)d_in[1];
    const int*   idx_br   = (const int*)d_in[2];
    const float* w_leaf   = (const float*)d_in[3];
    const float* b_leaf   = (const float*)d_in[4];
    const float* w_int    = (const float*)d_in[5];
    const float* b_int    = (const float*)d_in[6];
    const float* w_brleaf = (const float*)d_in[7];
    const float* b_brleaf = (const float*)d_in[8];
    const float* w_br     = (const float*)d_in[9];
    const float* b_br     = (const float*)d_in[10];
    const float* w_s      = (const float*)d_in[11];
    const float* b_s      = (const float*)d_in[12];
    const float* w_v      = (const float*)d_in[13];
    const float* b_v      = (const float*)d_in[14];

    char* ws = (char*)d_ws;
    const long xg_elts = (long)B_ * NB_ * TROWS * XP;    // ~51.7M bf16
    size_t off = 0;
    __bf16* xg    = (__bf16*)(ws + off); off += (size_t)xg_elts * 2;
    float*  feats = (float*) (ws + off); off += (size_t)B_ * 64 * T_ * 4;
    __bf16* pwL   = (__bf16*)(ws + off); off += (size_t)8 * KW * 4 * 512 * 2;
    __bf16* pwI   = (__bf16*)(ws + off); off += (size_t)4 * KW * 1 * 512 * 2;
    __bf16* pwH   = (__bf16*)(ws + off); off += (size_t)4 * 8 * KW * 4 * 512 * 2;
    __bf16* pwB   = (__bf16*)(ws + off); off += (size_t)4 * KW * 5 * 512 * 2;

    // weight fragment packing (f32 -> bf16 A-fragments)
    {
        int t1 = 8 * 1 * KW * 4 * 32;
        k_pack<<<(t1 + 255) / 256, 256, 0, stream>>>(w_leaf,   pwL, 8, 16,  128, 4);
        int t2 = 4 * 1 * KW * 1 * 32;
        k_pack<<<(t2 + 255) / 256, 256, 0, stream>>>(w_int,    pwI, 4, 16,  32,  1);
        int t3 = 4 * 8 * KW * 4 * 32;
        k_pack<<<(t3 + 255) / 256, 256, 0, stream>>>(w_brleaf, pwH, 4, 128, 128, 4);
        int t4 = 4 * 1 * KW * 5 * 32;
        k_pack<<<(t4 + 255) / 256, 256, 0, stream>>>(w_br,     pwB, 4, 16,  144, 5);
    }

    // gather/permute/convert x
    unsigned gblocks = (unsigned)((xg_elts + 255) / 256);
    k_gather<<<gblocks, 256, 0, stream>>>(x, idx_leaf, idx_br, xg, xg_elts);

    // fused branch compute (186 KB dynamic LDS per workgroup)
    (void)hipFuncSetAttribute((const void*)k_branch,
                              hipFuncAttributeMaxDynamicSharedMemorySize,
                              SMEM_BYTES);
    k_branch<<<B_ * NB_ * (T_ / TB), 256, SMEM_BYTES, stream>>>(
        xg, pwL, b_leaf, pwI, b_int, pwH, b_brleaf, pwB, b_br, feats);

    // head convs
    k_head<<<(B_ * T_ + 255) / 256, 256, 0, stream>>>(feats, w_s, b_s, w_v, b_v,
                                                      (float*)d_out);
}
